// Decoder_lstm_84198538871343
// MI455X (gfx1250) — matrix operand order repaired and checked by reading, not compiled, not run on hardware
//
#include <hip/hip_runtime.h>
#include <hip/hip_fp16.h>
#include <math.h>

// ---------------------------------------------------------------------------
// LSTM caption decoder for MI455X (gfx1250, wave32, WMMA).
// 16 persistent blocks x 512 threads (16 waves). Each block owns 16 batch
// rows for the whole 255-step recurrence (rows independent -> no grid sync).
// Each wave owns a 16-wide hidden slice and keeps i/f/g/o fragments plus
// c,h in registers; h is republished per step into an 8KB LDS A-operand
// buffer. Weights are pre-packed into WMMA B-operand layout (f16) and
// streamed from L2 every step. An opaque *integer offset* (not pointer!)
// is laundered through inline asm per iteration: this defeats LICM (which
// otherwise hoists ~470 VGPRs of weight loads and spills) while keeping
// address-space inference intact so loads stay global_load_b128, not flat.
// ---------------------------------------------------------------------------

typedef __attribute__((ext_vector_type(16))) _Float16 v16h;
typedef __attribute__((ext_vector_type(8)))  float    v8f;

#define BS      256
#define TT      256
#define STEPS   255
#define HID     256
#define INP     64
#define WV      100
#define VOCAB   600
#define VOCAB_P 608
#define NT_G    64           // gate N tiles (1024/16)
#define KC_G    10           // gate K chunks ((64+256)/32), kc 0,1 = x part
#define NT_O    38           // logits N tiles (608/16)
#define KC_O    8            // logits K chunks (256/32)

// ---- workspace layout (bytes) ----
#define OFF_SORT   0u
#define OFF_DECLEN 1024u
#define OFF_H      2048u
#define OFF_C      (OFF_H  + BS*HID*4u)                       // 264192
#define OFF_BG     (OFF_C  + BS*HID*4u)                       // 526336
#define OFF_BO     (OFF_BG + 4096u)                           // 530432
#define OFF_WG     (OFF_BO + 2560u)                           // 532992 (32B aligned)
#define OFF_WO     (OFF_WG + (unsigned)(NT_G*KC_G*512*2))     // 1188352
#define OFF_XP     (OFF_WO + (unsigned)(NT_O*KC_O*512*2))     // 1499648
// xpack size: 16*255*2*512*2 = 8,355,840 B ; total ws ~ 9.86 MB

// ---- output layout (floats) ----
#define OUT_PRED_N ((size_t)BS*STEPS*VOCAB)                   // 39,168,000
#define OUT_CAPS_N ((size_t)BS*TT*WV)                         //  6,553,600

__device__ __forceinline__ float sigm(float x) { return 1.0f / (1.0f + __expf(-x)); }

__device__ __forceinline__ float tanh_fast(float x) {
#if __has_builtin(__builtin_amdgcn_tanhf)
  return __builtin_amdgcn_tanhf(x);           // gfx1250 v_tanh_f32
#elif __has_builtin(__builtin_amdgcn_tanh_f32)
  return __builtin_amdgcn_tanh_f32(x);
#else
  return tanhf(x);
#endif
}

__device__ __forceinline__ v8f wmma16(v16h a, v16h b, v8f c) {
  // v_wmma_f32_16x16x32_f16 : D = A(16x32 f16) * B(32x16 f16) + C(16x16 f32)
  return __builtin_amdgcn_wmma_f32_16x16x32_f16(
      /*neg_a=*/false, a, /*neg_b=*/false, b,
      /*c_mod=*/(short)0, c, /*reuse_a=*/false, /*reuse_b=*/false);
}

__device__ __forceinline__ v8f splat8(float b) {
  v8f v = {b, b, b, b, b, b, b, b};
  return v;
}

// A-operand lane mapping for 16-bit 16x32 tiles (contiguous, matches B pack):
//  lane<16 : elements 0..15 -> K 0..15
//  lane>=16: elements 0..15 -> K 16..31
// inverse: given koff (K%32) -> (lane-half, element j)
__device__ __forceinline__ void a_invmap(int koff, int& loff, int& j) {
  loff = (koff < 16) ? 0 : 16;
  j = koff & 15;
}

// ---------------------------------------------------------------------------
// 1) stable descending sort of caption lengths (rank counting, bs=256)
// ---------------------------------------------------------------------------
__global__ void k_sort(const int* __restrict__ len, int* __restrict__ sidx,
                       int* __restrict__ dlen, float* __restrict__ out_dlen,
                       float* __restrict__ out_sidx) {
  __shared__ int L[BS];
  int i = threadIdx.x;
  L[i] = len[i];
  __syncthreads();
  int li = L[i], r = 0;
  for (int j = 0; j < BS; ++j) {
    int lj = L[j];
    r += (lj > li) || (lj == li && j < i);
  }
  sidx[r] = i;
  dlen[r] = li - 1;
  out_sidx[r] = (float)i;
  out_dlen[r] = (float)(li - 1);
}

// ---------------------------------------------------------------------------
// 2) combined gate bias (b_ih+b_hh) and padded output bias
// ---------------------------------------------------------------------------
__global__ void k_bias(const float* __restrict__ b_ih, const float* __restrict__ b_hh,
                       const float* __restrict__ b_out, float* __restrict__ bg,
                       float* __restrict__ bo) {
  int i = blockIdx.x * 256 + threadIdx.x;
  if (i < 4 * HID) {
    bg[i] = b_ih[i] + b_hh[i];
  } else {
    int n = i - 4 * HID;
    if (n < VOCAB_P) bo[n] = (n < VOCAB) ? b_out[n] : 0.0f;
  }
}

// ---------------------------------------------------------------------------
// 3) pack [w_ih | w_hh] (transposed) into WMMA B-operand order, f16.
//    B layout: lane l -> column N = 16*tile + (l&15),
//              K = 32*kc + (l<16 ? 0 : 16) + j, j = 0..15 contiguous.
// ---------------------------------------------------------------------------
__global__ void k_pack_wg(const float* __restrict__ w_ih, const float* __restrict__ w_hh,
                          _Float16* __restrict__ Wg) {
  int id = blockIdx.x * 256 + threadIdx.x;      // < 64*10*512
  int j    = id & 15;
  int l    = (id >> 4) & 31;
  int kc   = (id >> 9) % KC_G;
  int tile = id / (KC_G * 512);
  int n = tile * 16 + (l & 15);
  int k = kc * 32 + ((l < 16) ? 0 : 16) + j;    // combined K: 0..63 = x, 64..319 = h
  float v = (k < INP) ? w_ih[n * INP + k] : w_hh[n * HID + (k - INP)];
  Wg[id] = (_Float16)v;
}

__global__ void k_pack_wo(const float* __restrict__ W_out, _Float16* __restrict__ Wo) {
  int id = blockIdx.x * 256 + threadIdx.x;      // < 38*8*512
  int j    = id & 15;
  int l    = (id >> 4) & 31;
  int kc   = (id >> 9) & 7;
  int tile = id >> 12;
  int n = tile * 16 + (l & 15);
  int k = kc * 32 + ((l < 16) ? 0 : 16) + j;
  float v = (n < VOCAB) ? W_out[n * HID + k] : 0.0f;
  Wo[id] = (_Float16)v;
}

// ---------------------------------------------------------------------------
// 4) h0/c0 from gathered encoder rows
// ---------------------------------------------------------------------------
__global__ void k_init_state(const float* __restrict__ enc, const float* __restrict__ W_h,
                             const float* __restrict__ b_h, const float* __restrict__ W_c,
                             const float* __restrict__ b_c, const int* __restrict__ sidx,
                             float* __restrict__ hs, float* __restrict__ cs) {
  int m = blockIdx.x, n = threadIdx.x;
  int src = sidx[m];
  const float* e  = enc + (size_t)src * INP;
  const float* wh = W_h + (size_t)n * INP;
  const float* wc = W_c + (size_t)n * INP;
  float ah = b_h[n], ac = b_c[n];
  for (int k = 0; k < INP; ++k) { float ev = e[k]; ah += ev * wh[k]; ac += ev * wc[k]; }
  hs[(size_t)m * HID + n] = ah;
  cs[(size_t)m * HID + n] = ac;
}

// ---------------------------------------------------------------------------
// 5) gather caps (output #2), embedding GEMM, pack x into WMMA A-layout (f16)
//    xpack layout: [blk(16)][t(255)][kc(2)][lane(32)][j(16)] halves
// ---------------------------------------------------------------------------
__global__ void k_emb(const float* __restrict__ caps, const float* __restrict__ W_emb,
                      const float* __restrict__ b_emb, const int* __restrict__ sidx,
                      float* __restrict__ out_caps, _Float16* __restrict__ xpack) {
  __shared__ float row[WV];
  int t = blockIdx.x, m = blockIdx.y, tid = threadIdx.x;
  int src = sidx[m];
  if (tid < WV) {
    float v = caps[((size_t)src * TT + t) * WV + tid];
    row[tid] = v;
    out_caps[((size_t)m * TT + t) * WV + tid] = v;
  }
  __syncthreads();
  if (t < STEPS && tid < INP) {
    const float* w = W_emb + (size_t)tid * WV;
    float a = b_emb[tid];
    for (int c = 0; c < WV; ++c) a += row[c] * w[c];
    int blk = m >> 4, r = m & 15;
    int kc = tid >> 5, koff = tid & 31;
    int loff, j;
    a_invmap(koff, loff, j);
    int lane = r + loff;
    size_t idx = ((((size_t)blk * STEPS + t) * 2 + kc) * 32 + lane) * 16 + j;
    xpack[idx] = (_Float16)a;
  }
}

// ---------------------------------------------------------------------------
// 6) persistent LSTM kernel: 16 blocks x 512 threads (16 waves)
// ---------------------------------------------------------------------------
__global__ void __launch_bounds__(512, 1)
k_lstm(const _Float16* __restrict__ Wg, const _Float16* __restrict__ Wo,
       const float* __restrict__ bias_g, const float* __restrict__ bias_o,
       const _Float16* __restrict__ xpack, const float* __restrict__ h0,
       const float* __restrict__ c0, const int* __restrict__ dlen,
       float* __restrict__ pred) {
  __shared__ __align__(32) _Float16 hpack[KC_O][32][16];  // 8KB, h in A-layout
  __shared__ float bg[4 * HID];
  __shared__ float bo[VOCAB_P];
  __shared__ int   dl[16];

  const int tid   = threadIdx.x;
  const int wave  = tid >> 5;          // 0..15
  const int lane  = tid & 31;
  const int blk   = blockIdx.x;        // 0..15
  const int col   = lane & 15;
  const int mbase = (lane >> 4) * 8;   // C/D fragment row base
  const int nloc  = wave * 16 + col;   // this wave's hidden column (0..255)
  const int laneoff = lane * 16;

  for (int i = tid; i < 4 * HID; i += 512) bg[i] = bias_g[i];
  for (int i = tid; i < VOCAB_P; i += 512) bo[i] = bias_o[i];
  if (tid < 16) dl[tid] = dlen[blk * 16 + tid];

  // persistent per-wave cell-state fragments (rows mbase..mbase+7, col nloc)
  v8f hreg, creg;
#pragma unroll
  for (int r = 0; r < 8; ++r) {
    int m = mbase + r;
    hreg[r] = h0[((size_t)blk * 16 + m) * HID + nloc];
    creg[r] = c0[((size_t)blk * 16 + m) * HID + nloc];
  }

  // precompute this wave/lane's slot in the A-layout h buffer
  int loff, jn;
  a_invmap(nloc & 31, loff, jn);
  const int kcn = nloc >> 5;

  __syncthreads();                                  // biases/dl visible
#pragma unroll
  for (int r = 0; r < 8; ++r)
    hpack[kcn][(mbase + r) + loff][jn] = (_Float16)hreg[r];

  const float bgi = bg[nloc], bgf = bg[nloc + 256];
  const float bgg = bg[nloc + 512], bgo = bg[nloc + 768];
  const float bo0 = bo[nloc];                 // logits tile "wave"    (n =   0..255)
  const float bo1 = bo[nloc + 256];           // logits tile "wave+16" (n = 256..511)
  const float bo2 = (wave < 6) ? bo[nloc + 512] : 0.0f;  // tail tile (n = 512..607)
  __syncthreads();                                  // initial hpack visible

  // Opaque zero offsets, re-laundered every iteration: blocks LICM of the
  // weight loads (register-pressure bomb) without touching the pointers, so
  // address-space inference keeps the loads on the GLOBAL path (split
  // LOADcnt/DScnt counters, saddr+voffset addressing) instead of FLAT.
  size_t wg_ofs = 0, wo_ofs = 0;

  for (int t = 0; t < STEPS; ++t) {
    asm volatile("" : "+s"(wg_ofs), "+s"(wo_ofs));
    const _Float16* wg_t = Wg + wg_ofs;
    const _Float16* wo_t = Wo + wo_ofs;

    // ---- A operands for the x part (precomputed f16, A-layout, from L2)
    const _Float16* xp = xpack + ((size_t)(blk * STEPS + t)) * 1024 + laneoff;
    v16h ax0 = *(const v16h*)(xp);
    v16h ax1 = *(const v16h*)(xp + 512);

    // this wave's four gate-tile weight streams (i, f, g, o)
    const _Float16* wp0 = wg_t + ((size_t)(wave)      * KC_G) * 512 + laneoff;
    const _Float16* wp1 = wg_t + ((size_t)(wave + 16) * KC_G) * 512 + laneoff;
    const _Float16* wp2 = wg_t + ((size_t)(wave + 32) * KC_G) * 512 + laneoff;
    const _Float16* wp3 = wg_t + ((size_t)(wave + 48) * KC_G) * 512 + laneoff;

    // warm logits weights while we do the gates GEMM (global_prefetch_b8)
    __builtin_prefetch(wo_t + ((size_t)wave * KC_O) * 512 + laneoff, 0, 1);
    __builtin_prefetch(wo_t + ((size_t)(wave + 16) * KC_O) * 512 + laneoff, 0, 1);
    if (wave < 6)
      __builtin_prefetch(wo_t + ((size_t)(wave + 32) * KC_O) * 512 + laneoff, 0, 1);

    // ---- gates GEMM: kc-outer, gate-inner (shared A, 4 independent chains)
    v8f a0 = splat8(bgi), a1 = splat8(bgf), a2 = splat8(bgg), a3 = splat8(bgo);
    a0 = wmma16(ax0, *(const v16h*)(wp0), a0);
    a1 = wmma16(ax0, *(const v16h*)(wp1), a1);
    a2 = wmma16(ax0, *(const v16h*)(wp2), a2);
    a3 = wmma16(ax0, *(const v16h*)(wp3), a3);
    a0 = wmma16(ax1, *(const v16h*)(wp0 + 512), a0);
    a1 = wmma16(ax1, *(const v16h*)(wp1 + 512), a1);
    a2 = wmma16(ax1, *(const v16h*)(wp2 + 512), a2);
    a3 = wmma16(ax1, *(const v16h*)(wp3 + 512), a3);
#pragma unroll 2
    for (int kc = 0; kc < KC_O; ++kc) {
      v16h ah = *(const v16h*)&hpack[kc][lane][0];
      size_t o = (size_t)(2 + kc) * 512;
      a0 = wmma16(ah, *(const v16h*)(wp0 + o), a0);
      a1 = wmma16(ah, *(const v16h*)(wp1 + o), a1);
      a2 = wmma16(ah, *(const v16h*)(wp2 + o), a2);
      a3 = wmma16(ah, *(const v16h*)(wp3 + o), a3);
    }

    __syncthreads();  // all waves done reading old hpack

    // ---- in-register LSTM cell update (fp32 state, length masking)
#pragma unroll
    for (int r = 0; r < 8; ++r) {
      float i_ = sigm(a0[r]);
      float f_ = sigm(a1[r]);
      float g_ = tanh_fast(a2[r]);
      float o_ = sigm(a3[r]);
      float cn = f_ * creg[r] + i_ * g_;
      float hn = o_ * tanh_fast(cn);
      bool on = t < dl[mbase + r];
      creg[r] = on ? cn : creg[r];
      hreg[r] = on ? hn : hreg[r];
    }
#pragma unroll
    for (int r = 0; r < 8; ++r)
      hpack[kcn][(mbase + r) + loff][jn] = (_Float16)hreg[r];

    __syncthreads();  // new hpack visible

    // ---- logits: tiles (wave) and (wave+16) interleaved, + tail (wave+32)
    {
      const _Float16* q0 = wo_t + ((size_t)wave * KC_O) * 512 + laneoff;
      const _Float16* q1 = wo_t + ((size_t)(wave + 16) * KC_O) * 512 + laneoff;
      v8f l0 = splat8(bo0), l1 = splat8(bo1);
#pragma unroll 2
      for (int kc = 0; kc < KC_O; ++kc) {
        v16h ah = *(const v16h*)&hpack[kc][lane][0];
        l0 = wmma16(ah, *(const v16h*)(q0 + (size_t)kc * 512), l0);
        l1 = wmma16(ah, *(const v16h*)(q1 + (size_t)kc * 512), l1);
      }
      int n0 = nloc;                    // < 256, always valid
      int n1 = nloc + 256;              // < 512, always valid
#pragma unroll
      for (int r = 0; r < 8; ++r) {
        int m = mbase + r;
        bool on = t < dl[m];
        size_t base = ((size_t)(blk * 16 + m) * STEPS + t) * VOCAB;
        pred[base + n0] = on ? l0[r] : -1.0f;
        pred[base + n1] = on ? l1[r] : -1.0f;
      }
      if (wave < 6) {                   // tail tiles 32..37 (n = 512..607)
        const _Float16* q2 = wo_t + ((size_t)(wave + 32) * KC_O) * 512 + laneoff;
        v8f l2 = splat8(bo2);
#pragma unroll 2
        for (int kc = 0; kc < KC_O; ++kc) {
          v16h ah = *(const v16h*)&hpack[kc][lane][0];
          l2 = wmma16(ah, *(const v16h*)(q2 + (size_t)kc * 512), l2);
        }
        int n2 = nloc + 512;
        if (n2 < VOCAB) {
#pragma unroll
          for (int r = 0; r < 8; ++r) {
            int m = mbase + r;
            float v = (t < dl[m]) ? l2[r] : -1.0f;
            pred[((size_t)(blk * 16 + m) * STEPS + t) * VOCAB + n2] = v;
          }
        }
      }
    }
    __syncthreads();
  }
}

// ---------------------------------------------------------------------------
extern "C" void kernel_launch(void* const* d_in, const int* in_sizes, int n_in,
                              void* d_out, int out_size, void* d_ws, size_t ws_size,
                              hipStream_t stream) {
  (void)in_sizes; (void)n_in; (void)out_size; (void)ws_size;
  const float* enc   = (const float*)d_in[0];
  const float* caps  = (const float*)d_in[1];
  const float* w_ih  = (const float*)d_in[2];
  const float* w_hh  = (const float*)d_in[3];
  const float* b_ih  = (const float*)d_in[4];
  const float* b_hh  = (const float*)d_in[5];
  const float* W_out = (const float*)d_in[6];
  const float* b_out = (const float*)d_in[7];
  const float* W_emb = (const float*)d_in[8];
  const float* b_emb = (const float*)d_in[9];
  const float* W_h   = (const float*)d_in[10];
  const float* b_h   = (const float*)d_in[11];
  const float* W_c   = (const float*)d_in[12];
  const float* b_c   = (const float*)d_in[13];
  const int*   lens  = (const int*)d_in[14];

  char* ws = (char*)d_ws;
  int*      sidx   = (int*)(ws + OFF_SORT);
  int*      dlen   = (int*)(ws + OFF_DECLEN);
  float*    hstate = (float*)(ws + OFF_H);
  float*    cstate = (float*)(ws + OFF_C);
  float*    bg     = (float*)(ws + OFF_BG);
  float*    bo     = (float*)(ws + OFF_BO);
  _Float16* Wg     = (_Float16*)(ws + OFF_WG);
  _Float16* Wo     = (_Float16*)(ws + OFF_WO);
  _Float16* xpack  = (_Float16*)(ws + OFF_XP);

  float* out      = (float*)d_out;
  float* out_pred = out;
  float* out_caps = out + OUT_PRED_N;
  float* out_dlen = out + OUT_PRED_N + OUT_CAPS_N;
  float* out_sidx = out_dlen + BS;

  k_sort<<<1, 256, 0, stream>>>(lens, sidx, dlen, out_dlen, out_sidx);
  k_bias<<<7, 256, 0, stream>>>(b_ih, b_hh, b_out, bg, bo);
  k_pack_wg<<<NT_G * KC_G * 512 / 256, 256, 0, stream>>>(w_ih, w_hh, Wg);
  k_pack_wo<<<NT_O * KC_O * 512 / 256, 256, 0, stream>>>(W_out, Wo);
  k_init_state<<<BS, HID, 0, stream>>>(enc, W_h, b_h, W_c, b_c, sidx, hstate, cstate);
  k_emb<<<dim3(TT, BS), 128, 0, stream>>>(caps, W_emb, b_emb, sidx, out_caps, xpack);
  k_lstm<<<16, 512, 0, stream>>>(Wg, Wo, bg, bo, xpack, hstate, cstate, dlen, out_pred);
}
